// ParCUnit_29987461660753
// MI455X (gfx1250) — matrix operand order repaired and checked by reading, not compiled, not run on hardware
//
#include <hip/hip_runtime.h>
#include <hip/hip_bf16.h>

// ---------------------------------------------------------------------------
// out[b,co,h,w] = sum_{ci,j} wk[co,ci,j] * (X+pe)[b,ci,h,(w+j) mod 64] + bias[co]
// GEMM form with rotation folded into B row indices (A is w-independent):
//   out[co, n, w] = sum_{ci,j} Apack[co, ci*64+j] * Xt[ci*64 + ((j+w)&63)][n]
// Apack: 13->64 interpolated kernel, pre-packed in global memory in the exact
//        CDNA5 WMMA 16-bit A-fragment layout -> each K-chunk (one ci) is a
//        contiguous 32KB block, DMA'd to LDS with tensor_load_to_lds (TDM).
// Xt:    bf16(X + pe_up) transposed to [K=16384][N=2048].
// ---------------------------------------------------------------------------

typedef __attribute__((ext_vector_type(16))) __bf16          v16bf;
typedef __attribute__((ext_vector_type(8)))  float           v8f;
typedef __attribute__((ext_vector_type(8)))  unsigned short  ushort8;
typedef __attribute__((ext_vector_type(4)))  unsigned int    v4u;
typedef __attribute__((ext_vector_type(8)))  int             v8i;
typedef __attribute__((ext_vector_type(4)))  int             v4i;

#define CDIM   256
#define HDIM   64
#define WDIM   64
#define K0     13
#define NROW   2048            // B*H
#define KTOT   16384           // C*64
#define NT     128             // N-tile per workgroup
#define ACHUNK_DW 8192         // packed A chunk: 2kt*16mt*32lane*8v dwords = 32KB

// ---------------------------------------------------------------------------
// Prep 1: Xt[k=ci*64+u][n=b*64+h] = bf16( X[b,ci,h,u] + pe_up[ci,u] )
// ---------------------------------------------------------------------------
__global__ __launch_bounds__(256) void prep_xt(const float* __restrict__ X,
                                               const float* __restrict__ pe,
                                               __hip_bfloat16* __restrict__ Xt) {
    int idx = blockIdx.x * 256 + threadIdx.x;
    int n = idx & (NROW - 1);
    int k = idx >> 11;
    int b = n >> 6, h = n & 63;
    int ci = k >> 6, u = k & 63;

    float s = (u + 0.5f) * (13.0f / 64.0f) - 0.5f;
    s = fminf(fmaxf(s, 0.0f), 12.0f);
    int i0 = (int)s;
    float f = s - (float)i0;
    int i1 = min(i0 + 1, 12);
    float peu = pe[ci * K0 + i0] * (1.0f - f) + pe[ci * K0 + i1] * f;

    float xv = X[(((size_t)b * CDIM + ci) * HDIM + h) * WDIM + u] + peu;
    Xt[(size_t)k * NROW + n] = __float2bfloat16(xv);
}

// ---------------------------------------------------------------------------
// Prep 2: pack interpolated kernel straight into WMMA A-fragment layout:
// Apack half-index = ((((ci*2+kt)*16+mt)*32+lane)*8+v)*2 + half
// (ISA 16-bit A 16x32: lanes0-15 K pairs {0..7,16..23}, lanes16-31 {8..15,24..31})
// ---------------------------------------------------------------------------
__global__ __launch_bounds__(256) void prep_apack(const float* __restrict__ Wsrc,
                                                  unsigned short* __restrict__ Ap) {
    int idx = blockIdx.x * 256 + threadIdx.x;          // (co,ci,j)
    int j  = idx & 63;
    int ci = (idx >> 6) & (CDIM - 1);
    int co = idx >> 14;

    float s = (j + 0.5f) * (13.0f / 64.0f) - 0.5f;
    s = fminf(fmaxf(s, 0.0f), 12.0f);
    int i0 = (int)s;
    float f = s - (float)i0;
    int i1 = min(i0 + 1, 12);
    const float* wrow = Wsrc + ((size_t)co * CDIM + ci) * K0;
    float wv = wrow[i0] * (1.0f - f) + wrow[i1] * f;

    __hip_bfloat16 hv = __float2bfloat16(wv);
    unsigned short bits = *(const unsigned short*)&hv;

    int kt = j >> 5, kl = j & 31;
    int g  = kl >> 3;
    int v  = ((g & 2) ? 4 : 0) + ((kl & 7) >> 1);
    int ln = (co & 15) + ((g & 1) ? 16 : 0);
    int mt = (co >> 4) & 15;
    Ap[(size_t)(((((ci * 2 + kt) * 16 + mt) * 32 + ln) * 8 + v) * 2 + (kl & 1))] = bits;
}

// ---------------------------------------------------------------------------
// TDM: DMA one contiguous packed-A chunk (32KB) global -> LDS.
// D# group0: count=1, lds_addr, global_addr, type=2.  group1: data_size=2B,
// 1-D tile of 16384 elements.  Issued by wave 0 only; tracked with TENSORcnt.
// ---------------------------------------------------------------------------
__device__ __forceinline__ void tdm_load_chunk(const unsigned short* gsrc,
                                               unsigned int* ldsdst) {
#if __has_builtin(__builtin_amdgcn_tensor_load_to_lds)
    unsigned long long ga = (unsigned long long)(uintptr_t)gsrc;
    unsigned int lds_off = (unsigned int)(uintptr_t)ldsdst;   // flat low 32b = LDS offset
    v4u g0;
    g0.x = 1u;                                                // count=1, user mode
    g0.y = lds_off;                                           // lds_addr
    g0.z = (unsigned int)ga;                                  // global_addr[31:0]
    g0.w = (unsigned int)((ga >> 32) & 0x01FFFFFFu) | (2u << 30); // ga[56:32] | type=2
    v8i g1 = {0, 0, 0, 0, 0, 0, 0, 0};
    g1[0] = 0x00010000;            // wg_mask=0, data_size=1 (2 bytes)
    g1[1] = (int)(16384u << 16);   // tensor_dim0[15:0] @bits63:48  (=16384)
    g1[2] = (int)(1u << 16);       // tensor_dim0 hi=0; tensor_dim1[15:0]=1 @bits95:80
    g1[3] = (int)(16384u << 16);   // tile_dim0 @bits127:112 (=16384)
    g1[4] = 1;                     // tile_dim1=1, tile_dim2=0
    g1[5] = 16384;                 // tensor_dim0_stride[31:0]
    v4i z4 = {0, 0, 0, 0};
    v8i z8 = {0, 0, 0, 0, 0, 0, 0, 0};
    __builtin_amdgcn_tensor_load_to_lds(g0, g1, z4, z4, z8, 0);
#endif
}

__device__ __forceinline__ void tdm_wait() {
#if __has_builtin(__builtin_amdgcn_tensor_load_to_lds)
    __builtin_amdgcn_s_wait_tensorcnt(0);
#endif
}

// ---------------------------------------------------------------------------
// GEMM: grid = 64 (w) * 16 (n-tiles), block 512 (16 waves: 4 M x 4 N).
// Block tile M=256 (all co) x N=128; per-wave 4 m-tiles x 2 n-tiles; K-chunk=64.
// A chunk: TDM global->LDS (pre-packed fragments, double buffered).
// B chunk: vector loads of rotated Xt rows, even/odd-K rows paired into dwords.
// ---------------------------------------------------------------------------
__global__ __launch_bounds__(512, 1) void gemm_tdm(const unsigned short* __restrict__ Ap,
                                                   const __hip_bfloat16* __restrict__ Xt,
                                                   const float* __restrict__ bias,
                                                   float* __restrict__ out) {
    __shared__ unsigned int pA[2][ACHUNK_DW];      // 2 x 32KB packed A fragments
    __shared__ unsigned int pB[2][2 * 8 * 32 * 8]; // 2 x 16KB packed B fragments

    const int w    = blockIdx.x & 63;
    const int n0   = (blockIdx.x >> 6) * NT;
    const int tid  = threadIdx.x;
    const int lane = tid & 31;
    const int wave = tid >> 5;
    const int wm   = wave >> 2;
    const int wn   = wave & 3;

    v8f acc[4][2];
    #pragma unroll
    for (int i = 0; i < 4; ++i)
        #pragma unroll
        for (int j = 0; j < 2; ++j)
            acc[i][j] = (v8f){0.f, 0.f, 0.f, 0.f, 0.f, 0.f, 0.f, 0.f};

    // B fill: thread -> (K row-pair p, n-group ng).  u0=2p,u1=2p+1 share a
    // fragment dword (K=2v low half, K=2v+1 high half).  Rotation = row index.
    const int p   = tid >> 4;            // 0..31
    const int ng  = tid & 15;            // 0..15 (8 halves each)
    const int u0  = 2 * p;
    const int kt_b  = u0 >> 5;
    const int kl_b  = u0 & 31;
    const int lhi_b = (kl_b & 8) << 1;   // 0 or 16
    const int v_b   = ((kl_b & 16) >> 2) + ((kl_b & 7) >> 1);

    auto fillB = [&](int ci, int buf) {
        int r0 = (ci << 6) + ((u0 + w) & 63);
        int r1 = (ci << 6) + ((u0 + 1 + w) & 63);
        ushort8 a = *(const ushort8*)(Xt + (size_t)r0 * NROW + n0 + ng * 8);
        ushort8 b = *(const ushort8*)(Xt + (size_t)r1 * NROW + n0 + ng * 8);
        #pragma unroll
        for (int e = 0; e < 8; ++e) {
            int nl = ng * 8 + e;
            int nt = nl >> 4;
            int ln = (nl & 15) + lhi_b;
            pB[buf][((kt_b * 8 + nt) * 32 + ln) * 8 + v_b] =
                (unsigned int)a[e] | ((unsigned int)b[e] << 16);
        }
    };

#if !__has_builtin(__builtin_amdgcn_tensor_load_to_lds)
    // fallback: cooperative copy of the (already packed) A chunk
    auto fillA = [&](int ci, int buf) {
        const unsigned int* src = (const unsigned int*)Ap + (size_t)ci * ACHUNK_DW;
        #pragma unroll
        for (int q = 0; q < ACHUNK_DW / (512 * 4); ++q) {
            int i4 = (tid + 512 * q) * 4;
            *(v4u*)&pA[buf][i4] = *(const v4u*)&src[i4];
        }
    };
#endif

    // ---- prologue: chunk 0 into buffer 0 ----
#if __has_builtin(__builtin_amdgcn_tensor_load_to_lds)
    if (wave == 0) tdm_load_chunk(Ap, &pA[0][0]);
#else
    fillA(0, 0);
#endif
    fillB(0, 0);
#if __has_builtin(__builtin_amdgcn_tensor_load_to_lds)
    if (wave == 0) tdm_wait();
#endif
    __syncthreads();

    for (int ci = 0; ci < CDIM; ++ci) {
        const int cur = ci & 1;
        const int nxt = cur ^ 1;

        // ---- prefetch next chunk (overlaps with compute below) ----
        if (ci < CDIM - 1) {
#if __has_builtin(__builtin_amdgcn_tensor_load_to_lds)
            if (wave == 0)
                tdm_load_chunk(Ap + (size_t)(ci + 1) * ACHUNK_DW * 2, &pA[nxt][0]);
#else
            fillA(ci + 1, nxt);
#endif
            fillB(ci + 1, nxt);
        }

        // ---- compute: 2 k-steps x (4 m-tiles x 2 n-tiles) WMMA ----
        #pragma unroll
        for (int kt = 0; kt < 2; ++kt) {
            v16bf afrag[4], bfrag[2];
            #pragma unroll
            for (int i = 0; i < 4; ++i)
                afrag[i] = *(const v16bf*)&pA[cur][((kt * 16 + (wm * 4 + i)) * 32 + lane) * 8];
            #pragma unroll
            for (int j = 0; j < 2; ++j)
                bfrag[j] = *(const v16bf*)&pB[cur][((kt * 8 + (wn * 2 + j)) * 32 + lane) * 8];
            #pragma unroll
            for (int i = 0; i < 4; ++i)
                #pragma unroll
                for (int j = 0; j < 2; ++j)
                    acc[i][j] = __builtin_amdgcn_wmma_f32_16x16x32_bf16(
                        false, afrag[i], false, bfrag[j],
                        (short)0, acc[i][j], false, false);
        }

#if __has_builtin(__builtin_amdgcn_tensor_load_to_lds)
        if (wave == 0 && ci < CDIM - 1) tdm_wait();
#endif
        __syncthreads();
    }

    // ---- epilogue: D 16x16: lanes0-15 N=lane,M=r ; lanes16-31 N=lane-16,M=8+r ----
    const int nlane = lane & 15;
    const int mhi   = (lane >> 4) * 8;
    #pragma unroll
    for (int i = 0; i < 4; ++i) {
        #pragma unroll
        for (int j = 0; j < 2; ++j) {
            int n = n0 + wn * 32 + j * 16 + nlane;
            int b = n >> 6, h = n & 63;
            #pragma unroll
            for (int r = 0; r < 8; ++r) {
                int co = wm * 64 + i * 16 + mhi + r;
                out[(((size_t)b * CDIM + co) * HDIM + h) * WDIM + w] =
                    acc[i][j][r] + bias[co];
            }
        }
    }
}

// ---------------------------------------------------------------------------
extern "C" void kernel_launch(void* const* d_in, const int* in_sizes, int n_in,
                              void* d_out, int out_size, void* d_ws, size_t ws_size,
                              hipStream_t stream) {
    const float* X    = (const float*)d_in[0];
    const float* Wsrc = (const float*)d_in[1];
    const float* bias = (const float*)d_in[2];
    const float* pe   = (const float*)d_in[3];

    // workspace: Xt bf16 [16384][2048] = 64MB, then packed A = 8MB
    __hip_bfloat16* Xt  = (__hip_bfloat16*)d_ws;
    unsigned short* Ap  = (unsigned short*)((char*)d_ws + (size_t)KTOT * NROW * 2);

    prep_xt<<<(KTOT * NROW) / 256, 256, 0, stream>>>(X, pe, Xt);
    prep_apack<<<(CDIM * CDIM * 64) / 256, 256, 0, stream>>>(Wsrc, Ap);

    gemm_tdm<<<64 * 16, 512, 0, stream>>>(Ap, Xt, bias, (float*)d_out);
}
